// GQA_25185688224136
// MI455X (gfx1250) — matrix operand order repaired and checked by reading, not compiled, not run on hardware
//
#include <hip/hip_runtime.h>

// GQA attention block for MI455X (gfx1250).
// All matmuls via v_wmma_f32_16x16x32_bf16; GEMM A-tiles staged with
// global_load_async_to_lds_b128 (ASYNCcnt) + ping-pong LDS double buffering.

typedef __bf16 bf16_t;
typedef __attribute__((ext_vector_type(4)))  __bf16 v4bf;
typedef __attribute__((ext_vector_type(8)))  __bf16 v8bf;
typedef __attribute__((ext_vector_type(16))) __bf16 v16bf;
typedef __attribute__((ext_vector_type(8)))  float  v8f;

#define S_LEN 2048
#define HID   2048
#define NH    32
#define NKV   8
#define HD    128
#define QN    (NH * HD)    // 4096
#define KVN   (NKV * HD)   // 1024

#define BM 128
#define BN 128
#define BK 32

static __device__ __forceinline__ v16bf mk16(v8bf a, v8bf b) {
  return __builtin_shufflevector(a, b, 0,1,2,3,4,5,6,7,8,9,10,11,12,13,14,15);
}
static __device__ __forceinline__ v8f wmma_bf16(v16bf a, v16bf b, v8f c) {
  return __builtin_amdgcn_wmma_f32_16x16x32_bf16(false, a, false, b, (short)0, c, false, false);
}
static __device__ __forceinline__ void async_ld_b128x2(unsigned lds, const bf16_t* g) {
  // IOFFSET applies to both global and LDS addresses (ISA 08_async_tensor §4.4),
  // so one LDS base covers both 16B halves of the 32B-per-lane copy.
  asm volatile("global_load_async_to_lds_b128 %0, %1, off"           :: "v"(lds), "v"(g) : "memory");
  asm volatile("global_load_async_to_lds_b128 %0, %1, off offset:16" :: "v"(lds), "v"(g) : "memory");
}
static __device__ __forceinline__ void wait_async0() {
  asm volatile("s_wait_asynccnt 0x0" ::: "memory");
}

// ---------------------------------------------------------------- f32 -> bf16
__global__ void cvt_f32_bf16_kernel(const float* __restrict__ s,
                                    bf16_t* __restrict__ d, int n4) {
  int i = blockIdx.x * blockDim.x + threadIdx.x;
  if (i >= n4) return;
  float4 f = ((const float4*)s)[i];
  v4bf o;
  o[0] = (__bf16)f.x; o[1] = (__bf16)f.y; o[2] = (__bf16)f.z; o[3] = (__bf16)f.w;
  ((v4bf*)d)[i] = o;
}

// ------------------------------------------------- tiled WMMA GEMM (+ bias)
// C[M,N] = A[M,K] @ B[K,N] + bias ; bf16 inputs ; out f32 or bf16 (template).
// 256 threads = 8 waves; 128x128x32 tile; wave tile 32x64 (2x4 WMMAs);
// double-buffered LDS, async A staging, one barrier per K-step.
template <bool F32OUT>
__global__ __launch_bounds__(256)
void gemm_bf16_kernel(const bf16_t* __restrict__ A, const bf16_t* __restrict__ B,
                      const float* __restrict__ bias, void* __restrict__ Cout,
                      int N, int K)
{
  __shared__ __align__(16) bf16_t As[2][BM][BK];   // A tile, row-major
  __shared__ __align__(16) bf16_t Bs[2][BN][BK];   // B tile, N-major: Bs[.][n][k]

  const int tid  = threadIdx.x;
  const int lane = tid & 31;
  const int wave = tid >> 5;
  const int wm   = wave & 3;
  const int wn   = wave >> 2;
  const int hi   = lane >> 4;
  const int l16  = lane & 15;
  const int klo  = hi * 8;          // A-fragment K base
  const int kb   = hi * 16;         // unused; B fragments use klo
  const unsigned bm = blockIdx.y * BM;
  const unsigned bn = blockIdx.x * BN;

  // staging assignments (fixed per thread)
  const int ar  = tid >> 1;                 // A: row 0..127
  const int ac  = (tid & 1) * 16;          // A: col 0 or 16
  const int kk0 = (tid * 2) & 31;          // B: two (k, n8) groups
  const int nn0 = ((tid * 2) >> 5) * 8;
  const int kk1 = (tid * 2 + 1) & 31;
  const int nn1 = ((tid * 2 + 1) >> 5) * 8;
  const bf16_t* aptr  = A + (size_t)(bm + ar) * (unsigned)K + ac;
  const bf16_t* bptr0 = B + (size_t)kk0 * (unsigned)N + bn + nn0;
  const bf16_t* bptr1 = B + (size_t)kk1 * (unsigned)N + bn + nn1;
  const unsigned ldsA0 = (unsigned)(uintptr_t)&As[0][ar][ac];
  const unsigned ldsA1 = (unsigned)(uintptr_t)&As[1][ar][ac];

  const v8f vz = {0.f,0.f,0.f,0.f,0.f,0.f,0.f,0.f};
  v8f acc[2][4];
#pragma unroll
  for (int i = 0; i < 2; ++i)
#pragma unroll
    for (int j = 0; j < 4; ++j) acc[i][j] = vz;

  auto stage = [&](int buf, int k0) {
    async_ld_b128x2(buf ? ldsA1 : ldsA0, aptr + k0);
    v8bf b0 = *(const v8bf*)(bptr0 + (size_t)k0 * (unsigned)N);
    v8bf b1 = *(const v8bf*)(bptr1 + (size_t)k0 * (unsigned)N);
#pragma unroll
    for (int j = 0; j < 8; ++j) Bs[buf][nn0 + j][kk0] = b0[j];
#pragma unroll
    for (int j = 0; j < 8; ++j) Bs[buf][nn1 + j][kk1] = b1[j];
  };

  auto compute = [&](int buf) {
    v16bf af[2], bfr[4];
#pragma unroll
    for (int sm = 0; sm < 2; ++sm) {
      int row = wm * 32 + sm * 16 + l16;
      af[sm] = mk16(*(const v8bf*)&As[buf][row][klo],
                    *(const v8bf*)&As[buf][row][klo + 16]);
    }
#pragma unroll
    for (int sn = 0; sn < 4; ++sn) {
      int col = wn * 64 + sn * 16 + l16;
      bfr[sn] = mk16(*(const v8bf*)&Bs[buf][col][klo],
                     *(const v8bf*)&Bs[buf][col][klo + 16]);
    }
#pragma unroll
    for (int sm = 0; sm < 2; ++sm)
#pragma unroll
      for (int sn = 0; sn < 4; ++sn)
        acc[sm][sn] = wmma_bf16(af[sm], bfr[sn], acc[sm][sn]);
  };

  const int ntiles = K / BK;                // even (64 or 128)
  stage(0, 0);
  wait_async0();
  __syncthreads();
  for (int i = 0; i < ntiles; i += 2) {
    if (i + 1 < ntiles) stage(1, (i + 1) * BK);
    compute(0);
    wait_async0();
    __syncthreads();
    if (i + 2 < ntiles) stage(0, (i + 2) * BK);
    compute(1);
    wait_async0();
    __syncthreads();
  }

  float*  Cf = (float*)Cout;
  bf16_t* Cb = (bf16_t*)Cout;
#pragma unroll
  for (int sm = 0; sm < 2; ++sm)
#pragma unroll
    for (int sn = 0; sn < 4; ++sn) {
      const unsigned n   = bn + wn * 64 + sn * 16 + l16;
      const float    bia = bias[n];
      const unsigned base = (bm + wm * 32 + sm * 16 + hi * 8) * (unsigned)N + n;
#pragma unroll
      for (int r = 0; r < 8; ++r) {
        float v = acc[sm][sn][r] + bia;
        if constexpr (F32OUT) Cf[base + r * (unsigned)N] = v;
        else                  Cb[base + r * (unsigned)N] = (bf16_t)v;
      }
    }
}

// ----------------------------------------------------- flash attention (GQA)
// One wave per (head, 16-row q tile). 32-key blocks, online softmax, causal.
__global__ __launch_bounds__(32)
void attn_kernel(const bf16_t* __restrict__ Q, const bf16_t* __restrict__ Kg,
                 const bf16_t* __restrict__ Vg, bf16_t* __restrict__ O)
{
  __shared__ __align__(16) bf16_t Pt[16][32];    // P tile (C-layout -> A-layout)
  __shared__ __align__(16) bf16_t Vt[128][32];   // V block transposed: Vt[d][key]

  const int lane = threadIdx.x;
  const int hi   = lane >> 4;
  const int l16  = lane & 15;
  const int klo  = hi * 8;
  const int kb   = hi * 16;
  const int h    = blockIdx.y;
  const int kvh  = h >> 2;                       // GROUP = 4
  const int q0   = blockIdx.x * 16;

  v16bf qf[4];
  const bf16_t* qrow = Q + (unsigned)(q0 + l16) * QN + h * HD;
#pragma unroll
  for (int c = 0; c < 4; ++c)
    qf[c] = mk16(*(const v8bf*)(qrow + c * 32 + klo),
                 *(const v8bf*)(qrow + c * 32 + klo + 16));

  const v8f vz = {0.f,0.f,0.f,0.f,0.f,0.f,0.f,0.f};
  v8f oacc[8];
#pragma unroll
  for (int d = 0; d < 8; ++d) oacc[d] = vz;
  float mrow[8], lrow[8];
#pragma unroll
  for (int r = 0; r < 8; ++r) { mrow[r] = -3.0e38f; lrow[r] = 0.f; }

  const float scale = 0.08838834764831845f;      // 1/sqrt(128)

  for (int j0 = 0; j0 < q0 + 16; j0 += 32) {
    { // stage V transposed: lane owns key j0+lane
      const bf16_t* vrow = Vg + (unsigned)(j0 + lane) * KVN + kvh * HD;
#pragma unroll
      for (int d = 0; d < 128; d += 8) {
        v8bf v = *(const v8bf*)(vrow + d);
#pragma unroll
        for (int j = 0; j < 8; ++j) Vt[d + j][lane] = v[j];
      }
    }
    // S = Q @ K^T
    v8f sacc[2]; sacc[0] = vz; sacc[1] = vz;
#pragma unroll
    for (int sb = 0; sb < 2; ++sb) {
      const bf16_t* krow = Kg + (unsigned)(j0 + sb * 16 + l16) * KVN + kvh * HD;
#pragma unroll
      for (int c = 0; c < 4; ++c) {
        v16bf kf = mk16(*(const v8bf*)(krow + c * 32 + klo),
                        *(const v8bf*)(krow + c * 32 + klo + 16));
        sacc[sb] = wmma_bf16(qf[c], kf, sacc[sb]);
      }
    }
    // causal mask + online softmax (row m = q0+r+8*hi, key = j0+sb*16+l16)
#pragma unroll
    for (int r = 0; r < 8; ++r) {
      int qidx = q0 + r + hi * 8;
#pragma unroll
      for (int sb = 0; sb < 2; ++sb) {
        int kidx = j0 + sb * 16 + l16;
        float s = sacc[sb][r] * scale;
        sacc[sb][r] = (kidx <= qidx) ? s : -3.0e38f;
      }
      float v = fmaxf(sacc[0][r], sacc[1][r]);
#pragma unroll
      for (int off = 1; off < 16; off <<= 1)
        v = fmaxf(v, __shfl_xor(v, off, 32));
      float mnew  = fmaxf(mrow[r], v);
      float alpha = __expf(mrow[r] - mnew);
      mrow[r] = mnew;
      lrow[r] = lrow[r] * alpha;
#pragma unroll
      for (int d = 0; d < 8; ++d) oacc[d][r] = oacc[d][r] * alpha;
#pragma unroll
      for (int sb = 0; sb < 2; ++sb) {
        float p = __expf(sacc[sb][r] - mnew);
        lrow[r] += p;
        Pt[r + hi * 8][sb * 16 + l16] = (bf16_t)p;
      }
    }
    __syncthreads();
    // O += P @ V
    v16bf pf = mk16(*(const v8bf*)&Pt[l16][klo], *(const v8bf*)&Pt[l16][klo + 16]);
#pragma unroll
    for (int d = 0; d < 8; ++d) {
      v16bf vf = mk16(*(const v8bf*)&Vt[d * 16 + l16][klo],
                      *(const v8bf*)&Vt[d * 16 + l16][klo + 16]);
      oacc[d] = wmma_bf16(pf, vf, oacc[d]);
    }
    __syncthreads();
  }

#pragma unroll
  for (int r = 0; r < 8; ++r) {
    float t = lrow[r];
#pragma unroll
    for (int off = 1; off < 16; off <<= 1)
      t += __shfl_xor(t, off, 32);
    lrow[r] = 1.0f / t;
  }
  const unsigned obase = (unsigned)(q0 + hi * 8) * QN + (unsigned)h * HD + l16;
#pragma unroll
  for (int d = 0; d < 8; ++d)
#pragma unroll
    for (int r = 0; r < 8; ++r)
      O[obase + r * (unsigned)QN + d * 16] = (bf16_t)(oacc[d][r] * lrow[r]);
}

// --------------------------------------------------------------------- launch
extern "C" void kernel_launch(void* const* d_in, const int* in_sizes, int n_in,
                              void* d_out, int out_size, void* d_ws, size_t ws_size,
                              hipStream_t stream)
{
  const float* x  = (const float*)d_in[0];
  const float* wq = (const float*)d_in[2];
  const float* bq = (const float*)d_in[3];
  const float* wk = (const float*)d_in[4];
  const float* bk = (const float*)d_in[5];
  const float* wv = (const float*)d_in[6];
  const float* bv = (const float*)d_in[7];
  const float* wo = (const float*)d_in[8];
  const float* bo = (const float*)d_in[9];
  float* out = (float*)d_out;

  char* ws = (char*)d_ws;
  size_t off = 0;
  auto carve = [&](size_t elems) {
    bf16_t* p = (bf16_t*)(ws + off);
    off += elems * sizeof(bf16_t);
    return p;
  };
  bf16_t* xb  = carve((size_t)S_LEN * HID);
  bf16_t* wqb = carve((size_t)HID * QN);
  bf16_t* wkb = carve((size_t)HID * KVN);
  bf16_t* wvb = carve((size_t)HID * KVN);
  bf16_t* wob = carve((size_t)QN * HID);
  bf16_t* Qb  = carve((size_t)S_LEN * QN);
  bf16_t* Kb  = carve((size_t)S_LEN * KVN);
  bf16_t* Vb  = carve((size_t)S_LEN * KVN);
  bf16_t* Ab  = carve((size_t)S_LEN * QN);   // total ~92 MB, fits 192 MB L2

  auto cvt = [&](const float* s, bf16_t* d, size_t n) {
    int n4 = (int)(n / 4);
    cvt_f32_bf16_kernel<<<dim3((n4 + 255) / 256), dim3(256), 0, stream>>>(s, d, n4);
  };
  cvt(x,  xb,  (size_t)S_LEN * HID);
  cvt(wq, wqb, (size_t)HID * QN);
  cvt(wk, wkb, (size_t)HID * KVN);
  cvt(wv, wvb, (size_t)HID * KVN);
  cvt(wo, wob, (size_t)QN * HID);

  gemm_bf16_kernel<false><<<dim3(QN / BN,  S_LEN / BM), dim3(256), 0, stream>>>(
      xb, wqb, bq, Qb, QN, HID);
  gemm_bf16_kernel<false><<<dim3(KVN / BN, S_LEN / BM), dim3(256), 0, stream>>>(
      xb, wkb, bk, Kb, KVN, HID);
  gemm_bf16_kernel<false><<<dim3(KVN / BN, S_LEN / BM), dim3(256), 0, stream>>>(
      xb, wvb, bv, Vb, KVN, HID);

  attn_kernel<<<dim3(S_LEN / 16, NH), dim3(32), 0, stream>>>(Qb, Kb, Vb, Ab);

  gemm_bf16_kernel<true><<<dim3(HID / BN, S_LEN / BM), dim3(256), 0, stream>>>(
      Ab, wob, bo, out, HID, QN);

  (void)in_sizes; (void)n_in; (void)out_size; (void)ws_size;
}
